// LIIF_84731114816194
// MI455X (gfx1250) — matrix operand order repaired and checked by reading, not compiled, not run on hardware
//
#include <hip/hip_runtime.h>
#include <hip/hip_bf16.h>

typedef _Float16 f16;
typedef __attribute__((ext_vector_type(16))) _Float16 v16h;
typedef __attribute__((ext_vector_type(8)))  _Float16 v8h;
typedef __attribute__((ext_vector_type(8)))  float    v8f;
typedef __attribute__((ext_vector_type(4)))  int      v4i;

#define HH   192
#define WW   192
#define CC   64
#define QQ   16384
#define NEQ  65536      // 4 ensembles * Q rows of the MLP batch
#define INDIM 580
#define KPAD 608        // 580 padded to 19*32 for 16x16x32 WMMA K-steps
#define HID  256
#define LPAD 8          // LDS row padding (halves) -> conflict-free ds_load_b128

// -------- gfx1250 async global->LDS path (guarded; falls back cleanly) -----
#if defined(__has_builtin)
#if __has_builtin(__builtin_amdgcn_global_load_async_to_lds_b128)
#define USE_ASYNC_COPY 1
#endif
#endif

#define AS1 __attribute__((address_space(1)))
#define AS3 __attribute__((address_space(3)))

__device__ __forceinline__ void wait_async0() {
#if defined(__has_builtin) && __has_builtin(__builtin_amdgcn_s_wait_asynccnt)
  __builtin_amdgcn_s_wait_asynccnt(0);
#else
  asm volatile("s_wait_asynccnt 0" ::: "memory");
#endif
}

// ---------------------------------------------------------------------------
// Conv 3x3, 3->64, SAME. One thread per pixel, loops output channels.
// ---------------------------------------------------------------------------
__global__ void conv3x3_kernel(const float* __restrict__ inp,
                               const float* __restrict__ cw,
                               const float* __restrict__ cb,
                               float* __restrict__ feat) {
  int p = blockIdx.x * blockDim.x + threadIdx.x;
  if (p >= HH * WW) return;
  int y = p / WW, x = p % WW;
  float in[27];
  #pragma unroll
  for (int ci = 0; ci < 3; ++ci)
    #pragma unroll
    for (int ky = 0; ky < 3; ++ky)
      #pragma unroll
      for (int kx = 0; kx < 3; ++kx) {
        int yy = y + ky - 1, xx = x + kx - 1;
        in[ci * 9 + ky * 3 + kx] =
            (yy >= 0 && yy < HH && xx >= 0 && xx < WW) ? inp[(ci * HH + yy) * WW + xx] : 0.f;
      }
  for (int c = 0; c < CC; ++c) {
    const float* w = cw + c * 27;   // uniform across wave -> scalar loads
    float acc = cb[c];
    #pragma unroll
    for (int i = 0; i < 27; ++i) acc = fmaf(w[i], in[i], acc);
    feat[c * (HH * WW) + p] = acc;
  }
}

// ---------------------------------------------------------------------------
// Weight repack: w[din][dout] fp32 -> Wt[dout][kpad] fp16 (transposed, K-padded)
// so a WMMA B-fragment is 32 contiguous bytes per lane.
// ---------------------------------------------------------------------------
__global__ void pack_wt_kernel(const float* __restrict__ w, f16* __restrict__ wt,
                               int din, int dout, int kpad) {
  int idx = blockIdx.x * blockDim.x + threadIdx.x;
  if (idx >= dout * kpad) return;
  int n = idx / kpad, k = idx % kpad;
  float v = (k < din) ? w[k * dout + n] : 0.f;
  wt[idx] = (f16)v;
}

// ---------------------------------------------------------------------------
// Ensemble bilinear sampler. One block per (ensemble, query). Builds one
// fp16 row of the MLP input matrix X[NEQ][KPAD] and the ensemble area.
// ---------------------------------------------------------------------------
__global__ void sample_kernel(const float* __restrict__ feat,
                              const float* __restrict__ coord,
                              const float* __restrict__ cell,
                              f16* __restrict__ X,
                              float* __restrict__ areas) {
  const int eq = blockIdx.x;          // e*Q + q
  const int e  = eq >> 14;
  const int q  = eq & (QQ - 1);
  const float offy = (e < 2) ? -1.f : 1.f;
  const float offx = (e & 1) ? 1.f : -1.f;
  const float c0 = coord[2 * q + 0], c1 = coord[2 * q + 1];
  const float rad = 1.f / 192.f;
  float ey = c0 + offy * rad + 1e-8f;
  float ex = c1 + offx * rad + 1e-8f;
  ey = fminf(fmaxf(ey, -1.f + 1e-10f), 1.f - 1e-10f);
  ex = fminf(fmaxf(ex, -1.f + 1e-10f), 1.f - 1e-10f);
  // sc flips the last axis: grid-x <- ens[...,1], grid-y <- ens[...,0]
  const float xf = (ex + 1.f) * (WW * 0.5f) - 0.5f;
  const float yf = (ey + 1.f) * (HH * 0.5f) - 0.5f;
  const float x0 = floorf(xf), y0 = floorf(yf);

  float wgt[4]; int yi[4], xi[4];
  #pragma unroll
  for (int k = 0; k < 4; ++k) {
    float yc = y0 + (float)(k >> 1);
    float xc = x0 + (float)(k & 1);
    float wv = (1.f - fabsf(xf - xc)) * (1.f - fabsf(yf - yc));
    float vl = (xc >= 0.f && xc < (float)WW && yc >= 0.f && yc < (float)HH) ? 1.f : 0.f;
    wgt[k] = wv * vl;
    yi[k]  = (int)fminf(fmaxf(yc, 0.f), (float)(HH - 1));
    xi[k]  = (int)fminf(fmaxf(xc, 0.f), (float)(WW - 1));
  }

  const size_t row = (size_t)eq * KPAD;
  for (int col = threadIdx.x; col < KPAD; col += blockDim.x) {
    float out;
    if (col < INDIM - 4) {                       // 576 unfolded conv features
      int c = col / 9, t = col % 9;
      int dy = t / 3 - 1, dx = t % 3 - 1;        // feat_u via padded shift
      float acc = 0.f;
      #pragma unroll
      for (int k = 0; k < 4; ++k) {
        int yy = yi[k] + dy, xx = xi[k] + dx;
        float v = (yy >= 0 && yy < HH && xx >= 0 && xx < WW)
                      ? feat[(c * HH + yy) * WW + xx] : 0.f;
        acc = fmaf(wgt[k], v, acc);
      }
      out = acc;
    } else if (col < INDIM) {                    // rel_coord (2) + rel_cell (2)
      float qy = 0.f, qx = 0.f;
      #pragma unroll
      for (int k = 0; k < 4; ++k) {
        float ry = -1.f + 2.f * ((float)yi[k] + 0.5f) / (float)HH;
        float rx = -1.f + 2.f * ((float)xi[k] + 0.5f) / (float)WW;
        qy = fmaf(wgt[k], ry, qy);
        qx = fmaf(wgt[k], rx, qx);
      }
      float rel0 = (c0 - qy) * (float)HH;
      float rel1 = (c1 - qx) * (float)WW;
      if      (col == 576) out = rel0;
      else if (col == 577) out = rel1;
      else if (col == 578) out = cell[2 * q + 0] * (float)HH;
      else                 out = cell[2 * q + 1] * (float)WW;
      if (col == 576) areas[eq] = fabsf(rel0 * rel1) + 1e-9f;
    } else {
      out = 0.f;                                 // K padding
    }
    X[row + col] = (f16)out;
  }
}

// ---------------------------------------------------------------------------
// WMMA GEMM + bias + ReLU:  Y[M][256] = relu(A[M][K] * Wt^T + b)
// A row-major fp16 (lda=K), Wt stored as [N][K] fp16 (transposed weights).
// Block: 8 waves = 32 M-rows x 256 N.  Wave: 16 M x 64 N (4 accumulators).
// A tile (32 x K) is staged in LDS once per block via gfx1250 async copies
// (ASYNCcnt) and consumed by ds_load_b128 A-fragments; B streams from
// L2/WGP$ (weights are shared by all 2048 blocks).
// ---------------------------------------------------------------------------
__device__ __forceinline__ void store_tile(v8f acc, const float* __restrict__ bias,
                                           f16* __restrict__ Y, int row0, int nc) {
  float bv = bias[nc];
  #pragma unroll
  for (int v = 0; v < 8; ++v) {
    float r = fmaxf(acc[v] + bv, 0.f);
    Y[(size_t)(row0 + v) * HID + nc] = (f16)r;
  }
}

__global__ __launch_bounds__(256) void gemm_wmma_relu_kernel(
    const f16* __restrict__ A, const f16* __restrict__ Bt,
    const float* __restrict__ bias, f16* __restrict__ Y, int K) {
  __shared__ f16 As[32 * (KPAD + LPAD)];       // 38.5 KB, max-K layout
  const int ldl  = K + LPAD;                   // LDS row stride (halves)
  const int tid  = threadIdx.x;
  const int lane = tid & 31;
  const int wave = tid >> 5;
  const int lg   = lane >> 4;                  // half-wave group
  const int ln   = lane & 15;
  const int mbase = blockIdx.x * 32;
  const int nbase = (wave & 3) * 64;

  // ---- stage A tile (32 rows x K halves) into LDS ----
  const int cpr   = K >> 3;                    // 16B chunks per row
  const int total = 32 * cpr;
#ifdef USE_ASYNC_COPY
  for (int c = tid; c < total; c += 256) {
    int r = c / cpr, k8 = c - r * cpr;
    const f16* g = A + (size_t)(mbase + r) * K + k8 * 8;
    const f16* l = &As[r * ldl + k8 * 8];
    __builtin_amdgcn_global_load_async_to_lds_b128(
        (AS1 v4i*)(uintptr_t)g,
        (AS3 v4i*)(uint32_t)(uintptr_t)l,       // flat-LDS aperture: offset in addr[31:0]
        0, 0);
  }
  wait_async0();
#else
  for (int c = tid; c < total; c += 256) {
    int r = c / cpr, k8 = c - r * cpr;
    *reinterpret_cast<v8h*>(&As[r * ldl + k8 * 8]) =
        *reinterpret_cast<const v8h*>(A + (size_t)(mbase + r) * K + k8 * 8);
  }
#endif
  __syncthreads();

  v8f acc0 = {}, acc1 = {}, acc2 = {}, acc3 = {};

  // A fragment from LDS (contiguous k order, same as B): lane ln = row,
  // lg picks k 0..15 / 16..31
  const f16* arow = &As[((wave >> 2) * 16 + ln) * ldl + lg * 16];
  // B fragment (contiguous k order, same as A): lane ln = col, lg picks k 0..15 / 16..31
  const f16* b0p = Bt + (size_t)(nbase +  0 + ln) * K + lg * 16;
  const f16* b1p = Bt + (size_t)(nbase + 16 + ln) * K + lg * 16;
  const f16* b2p = Bt + (size_t)(nbase + 32 + ln) * K + lg * 16;
  const f16* b3p = Bt + (size_t)(nbase + 48 + ln) * K + lg * 16;

  for (int kb = 0; kb < K; kb += 32) {
    v8h alo = *reinterpret_cast<const v8h*>(arow + kb);
    v8h ahi = *reinterpret_cast<const v8h*>(arow + kb + 8);
    v16h a = __builtin_shufflevector(alo, ahi, 0,1,2,3,4,5,6,7,8,9,10,11,12,13,14,15);
    v16h b0 = *reinterpret_cast<const v16h*>(b0p + kb);
    v16h b1 = *reinterpret_cast<const v16h*>(b1p + kb);
    v16h b2 = *reinterpret_cast<const v16h*>(b2p + kb);
    v16h b3 = *reinterpret_cast<const v16h*>(b3p + kb);
    acc0 = __builtin_amdgcn_wmma_f32_16x16x32_f16(false, a, false, b0, (short)0, acc0, false, false);
    acc1 = __builtin_amdgcn_wmma_f32_16x16x32_f16(false, a, false, b1, (short)0, acc1, false, false);
    acc2 = __builtin_amdgcn_wmma_f32_16x16x32_f16(false, a, false, b2, (short)0, acc2, false, false);
    acc3 = __builtin_amdgcn_wmma_f32_16x16x32_f16(false, a, false, b3, (short)0, acc3, false, false);
  }

  const int row0 = mbase + (wave >> 2) * 16 + lg * 8;   // lanes 16-31 hold M=8..15
  store_tile(acc0, bias, Y, row0, nbase +  0 + ln);
  store_tile(acc1, bias, Y, row0, nbase + 16 + ln);
  store_tile(acc2, bias, Y, row0, nbase + 32 + ln);
  store_tile(acc3, bias, Y, row0, nbase + 48 + ln);
}

// ---------------------------------------------------------------------------
// Head: 256->3 fp32 dot, tanh*(1+eps), area-normalized ensemble sum.
// ---------------------------------------------------------------------------
__global__ void final_kernel(const f16* __restrict__ Y, const float* __restrict__ w4,
                             const float* __restrict__ b4, const float* __restrict__ areas,
                             float* __restrict__ out) {
  int q = blockIdx.x * blockDim.x + threadIdx.x;
  if (q >= QQ) return;
  float a0 = areas[0 * QQ + q], a1 = areas[1 * QQ + q];
  float a2 = areas[2 * QQ + q], a3 = areas[3 * QQ + q];
  float tot = a0 + a1 + a2 + a3;
  float wA[4] = { a3 / tot, a2 / tot, a1 / tot, a0 / tot };   // areas[::-1]
  float o0 = 0.f, o1 = 0.f, o2 = 0.f;
  for (int e = 0; e < 4; ++e) {
    const f16* yrow = Y + (size_t)(e * QQ + q) * HID;
    float p0 = b4[0], p1 = b4[1], p2 = b4[2];
    for (int k = 0; k < HID; ++k) {
      float yv = (float)yrow[k];
      p0 = fmaf(yv, w4[k * 3 + 0], p0);
      p1 = fmaf(yv, w4[k * 3 + 1], p1);
      p2 = fmaf(yv, w4[k * 3 + 2], p2);
    }
    const float s = 1.01f;
    float we = wA[e];
    o0 = fmaf(tanhf(p0) * s, we, o0);
    o1 = fmaf(tanhf(p1) * s, we, o1);
    o2 = fmaf(tanhf(p2) * s, we, o2);
  }
  out[q * 3 + 0] = o0; out[q * 3 + 1] = o1; out[q * 3 + 2] = o2;
}

// ---------------------------------------------------------------------------
extern "C" void kernel_launch(void* const* d_in, const int* in_sizes, int n_in,
                              void* d_out, int out_size, void* d_ws, size_t ws_size,
                              hipStream_t stream) {
  (void)in_sizes; (void)n_in; (void)out_size; (void)ws_size;
  const float* inp    = (const float*)d_in[0];
  const float* coord  = (const float*)d_in[1];
  const float* cell   = (const float*)d_in[2];
  const float* conv_w = (const float*)d_in[3];
  const float* conv_b = (const float*)d_in[4];
  const float* w0 = (const float*)d_in[5];  const float* b0 = (const float*)d_in[6];
  const float* w1 = (const float*)d_in[7];  const float* b1 = (const float*)d_in[8];
  const float* w2 = (const float*)d_in[9];  const float* b2 = (const float*)d_in[10];
  const float* w3 = (const float*)d_in[11]; const float* b3 = (const float*)d_in[12];
  const float* w4 = (const float*)d_in[13]; const float* b4 = (const float*)d_in[14];

  char* ws = (char*)d_ws;
  size_t off = 0;
  auto alloc = [&](size_t bytes) -> void* {
    void* p = ws + off;
    off = (off + bytes + 255) & ~(size_t)255;
    return p;
  };
  float* feat  = (float*)alloc((size_t)CC * HH * WW * sizeof(float));   // 9.4 MB
  f16*   X     = (f16*)  alloc((size_t)NEQ * KPAD * sizeof(f16));       // 79.7 MB
  f16*   Y1    = (f16*)  alloc((size_t)NEQ * HID  * sizeof(f16));       // 33.5 MB
  f16*   Y2    = (f16*)  alloc((size_t)NEQ * HID  * sizeof(f16));       // 33.5 MB
  float* areas = (float*)alloc((size_t)NEQ * sizeof(float));
  f16*   Wt0   = (f16*)  alloc((size_t)HID * KPAD * sizeof(f16));
  f16*   Wt1   = (f16*)  alloc((size_t)HID * HID  * sizeof(f16));
  f16*   Wt2   = (f16*)  alloc((size_t)HID * HID  * sizeof(f16));
  f16*   Wt3   = (f16*)  alloc((size_t)HID * HID  * sizeof(f16));

  pack_wt_kernel<<<(HID * KPAD + 255) / 256, 256, 0, stream>>>(w0, Wt0, INDIM, HID, KPAD);
  pack_wt_kernel<<<(HID * HID  + 255) / 256, 256, 0, stream>>>(w1, Wt1, HID, HID, HID);
  pack_wt_kernel<<<(HID * HID  + 255) / 256, 256, 0, stream>>>(w2, Wt2, HID, HID, HID);
  pack_wt_kernel<<<(HID * HID  + 255) / 256, 256, 0, stream>>>(w3, Wt3, HID, HID, HID);

  conv3x3_kernel<<<(HH * WW + 127) / 128, 128, 0, stream>>>(inp, conv_w, conv_b, feat);
  sample_kernel<<<NEQ, 128, 0, stream>>>(feat, coord, cell, X, areas);

  gemm_wmma_relu_kernel<<<NEQ / 32, 256, 0, stream>>>(X,  Wt0, b0, Y1, KPAD);
  gemm_wmma_relu_kernel<<<NEQ / 32, 256, 0, stream>>>(Y1, Wt1, b1, Y2, HID);
  gemm_wmma_relu_kernel<<<NEQ / 32, 256, 0, stream>>>(Y2, Wt2, b2, Y1, HID);
  gemm_wmma_relu_kernel<<<NEQ / 32, 256, 0, stream>>>(Y1, Wt3, b3, Y2, HID);

  final_kernel<<<(QQ + 255) / 256, 256, 0, stream>>>(Y2, w4, b4, areas, (float*)d_out);
}
